// MultiHeadSelfAttention_54563264528841
// MI455X (gfx1250) — matrix operand order repaired and checked by reading, not compiled, not run on hardware
//
#include <hip/hip_runtime.h>
#include <hip/hip_bf16.h>

#define B_  4
#define S_  2048
#define D_  512
#define H_  8
#define DK_ 64
#define HD_ (H_*DK_)   // 512

typedef __bf16 bf16;
typedef __attribute__((ext_vector_type(16))) __bf16 v16bf;
typedef __attribute__((ext_vector_type(8)))  float   v8f;
typedef __attribute__((ext_vector_type(4)))  unsigned int u32x4;

union BF16x16 { v16bf v; u32x4 q[2]; };

__device__ __forceinline__ u32x4 ld_b128(const void* p) {
  return *reinterpret_cast<const u32x4*>(p);
}

__device__ __forceinline__ v8f wmma_bf16(v16bf a, v16bf b, v8f c) {
  // 8 args: (neg_a, A, neg_b, B, c_mod, C, reuse_a, reuse_b)
  return __builtin_amdgcn_wmma_f32_16x16x32_bf16(false, a, false, b, (short)0, c,
                                                 false, false);
}

// CDNA5 async copy: global -> LDS, tracked by ASYNCcnt (CDNA5 ISA)
__device__ __forceinline__ void async_load_b128(unsigned int lds_off, const void* gaddr) {
  asm volatile("global_load_async_to_lds_b128 %0, %1, off"
               :: "v"(lds_off), "v"(gaddr) : "memory");
}
__device__ __forceinline__ void wait_async0() {
  asm volatile("s_wait_asynccnt 0" ::: "memory");
}
__device__ __forceinline__ unsigned int lds_off32(const void* p) {
  // flat shared address: [63:32]=aperture, [31:0]=LDS byte offset
  return (unsigned int)(unsigned long long)p;
}

// ---------------- prep kernels ----------------

__global__ void cvt_f32_bf16(const float* __restrict__ src, bf16* __restrict__ dst, int n) {
  int i = blockIdx.x * blockDim.x + threadIdx.x;
  if (i < n) dst[i] = (bf16)src[i];
}

// src[rows][cols] (row-major) -> dst[cols][rows] bf16
__global__ void transpose_cvt(const float* __restrict__ src, bf16* __restrict__ dst,
                              int rows, int cols) {
  int i = blockIdx.x * blockDim.x + threadIdx.x;
  if (i < rows * cols) {
    int r = i / cols, c = i % cols;
    dst[(size_t)c * rows + r] = (bf16)src[i];
  }
}

// ---------------- fused QKV projection ----------------
// grid: B*S/64 blocks, 128 threads (4 waves, 16 rows each)
__global__ void __launch_bounds__(128) qkv_proj(
    const bf16* __restrict__ Qb,    // [B*S, D]
    const bf16* __restrict__ Wqt,   // [HD, D]  (transposed)
    const bf16* __restrict__ Wkt,
    const bf16* __restrict__ Wvt,
    const float* __restrict__ bq,
    const float* __restrict__ bk,
    const float* __restrict__ bv,
    bf16* __restrict__ qbuf,        // [B*H, S, DK]
    bf16* __restrict__ kbuf,        // [B*H, S, DK]
    bf16* __restrict__ vt)          // [B*H, DK, S]
{
  const int lane   = threadIdx.x & 31;
  const int wave   = threadIdx.x >> 5;
  const int mbase  = (blockIdx.x * 4 + wave) * 16;
  const bool lo    = lane < 16;
  const int row_a  = lane & 15;
  const int klo    = lo ? 0 : 16;   // A-layout half select
  const int kbhalf = lo ? 0 : 16;   // B-layout half select
  const int nlane  = lane & 15;

  for (int nt = 0; nt < HD_ / 16; ++nt) {
    const int col = nt * 16 + nlane;
    v8f aq = {}, ak = {}, av = {};
    for (int kc = 0; kc < D_ / 32; ++kc) {
      const int kb = kc * 32;
      BF16x16 a;
      const bf16* arow = Qb + (size_t)(mbase + row_a) * D_ + kb;
      a.q[0] = ld_b128(arow + klo);
      a.q[1] = ld_b128(arow + 8 + klo);

      BF16x16 b;
      const bf16* pq = Wqt + (size_t)col * D_ + kb + kbhalf;
      b.q[0] = ld_b128(pq); b.q[1] = ld_b128(pq + 8);
      aq = wmma_bf16(a.v, b.v, aq);

      const bf16* pk = Wkt + (size_t)col * D_ + kb + kbhalf;
      b.q[0] = ld_b128(pk); b.q[1] = ld_b128(pk + 8);
      ak = wmma_bf16(a.v, b.v, ak);

      const bf16* pv = Wvt + (size_t)col * D_ + kb + kbhalf;
      b.q[0] = ld_b128(pv); b.q[1] = ld_b128(pv + 8);
      av = wmma_bf16(a.v, b.v, av);
    }
    const int h = col >> 6, dk = col & 63;
    const float bqc = bq[col], bkc = bk[col], bvc = bv[col];
#pragma unroll
    for (int r = 0; r < 8; ++r) {
      const int g    = mbase + r + (lo ? 0 : 8);
      const int bidx = g / S_, sidx = g % S_;
      const size_t bh = (size_t)bidx * H_ + h;
      qbuf[(bh * S_ + sidx) * DK_ + dk] = (bf16)(aq[r] + bqc);
      kbuf[(bh * S_ + sidx) * DK_ + dk] = (bf16)(ak[r] + bkc);
      vt[(bh * DK_ + dk) * S_ + sidx]   = (bf16)(av[r] + bvc);
    }
  }
}

// ---------------- attention ----------------
// grid: B*H*(S/64) blocks, 128 threads (4 waves, 64 query rows per block).
// K/V tiles for 32 keys staged into LDS once per block via async copies.
#define KT_STRIDE 72   // 32 rows [key][dk], padded (36 dwords -> conflict-spread)
#define VT_STRIDE 40   // 64 rows [dk][key], padded (20 dwords -> conflict-free)

__global__ void __launch_bounds__(128) attn(
    const bf16* __restrict__ qbuf,   // [B*H, S, DK]
    const bf16* __restrict__ kbuf,   // [B*H, S, DK]
    const bf16* __restrict__ vt,     // [B*H, DK, S]
    const float* __restrict__ mask,  // [B, S]
    bf16* __restrict__ ctx)          // [B*S, HD]
{
  const int lane  = threadIdx.x & 31;
  const int wave  = threadIdx.x >> 5;
  const bool lo   = lane < 16;
  const int row_a = lane & 15;
  const int klo   = lo ? 0 : 16;
  const int nlane = lane & 15;
  const int bh    = blockIdx.x / (S_ / 64);
  const int qblk  = (blockIdx.x % (S_ / 64)) * 64;
  const int qbase = qblk + wave * 16;             // this wave's 16 query rows
  const int bidx  = bh / H_, h = bh % H_;

  __shared__ __align__(16) bf16 ktile[32 * KT_STRIDE];   // [key][dk]
  __shared__ __align__(16) bf16 vtile[64 * VT_STRIDE];   // [dk][key]
  __shared__ __align__(16) bf16 plds[4][16 * 40];        // per-wave probs 16x32

  const bf16* kbase = kbuf + (size_t)bh * S_ * DK_;
  const bf16* vbase = vt   + (size_t)bh * DK_ * S_;

  BF16x16 a0, a1;   // q rows as A operands, dk chunks 0..31 / 32..63
  {
    const bf16* qrow = qbuf + ((size_t)bh * S_ + qbase + row_a) * DK_;
    a0.q[0] = ld_b128(qrow + klo);
    a0.q[1] = ld_b128(qrow + 8 + klo);
    a1.q[0] = ld_b128(qrow + 32 + klo);
    a1.q[1] = ld_b128(qrow + 40 + klo);
  }

  v8f cacc[4] = {{}, {}, {}, {}};
  float sums[8] = {0, 0, 0, 0, 0, 0, 0, 0};

  for (int kb = 0; kb < S_; kb += 32) {
    // ---- stage K (32x64) and V (64x32) tiles: 8 async b128 instrs each, 2/wave
#pragma unroll
    for (int i = 0; i < 2; ++i) {
      const int c = wave * 64 + i * 32 + lane;    // 0..255 16B-chunk index
      const int key = c >> 3, sub = c & 7;        // K tile: 8 chunks per 64-elem row
      async_load_b128(lds_off32(&ktile[key * KT_STRIDE + sub * 8]),
                      kbase + (size_t)(kb + key) * DK_ + sub * 8);
      const int dkr = c >> 2, sub2 = c & 3;       // V tile: 4 chunks per 32-elem row
      async_load_b128(lds_off32(&vtile[dkr * VT_STRIDE + sub2 * 8]),
                      vbase + (size_t)dkr * S_ + kb + sub2 * 8);
    }
    wait_async0();
    __syncthreads();

    // ---- scores: 16 queries x 32 keys
#pragma unroll
    for (int j2 = 0; j2 < 2; ++j2) {
      const int keyl = j2 * 16 + nlane;           // this lane's key column in tile
      BF16x16 bm;
      const bf16* kl0 = &ktile[keyl * KT_STRIDE + (lo ? 0 : 16)];
      bm.q[0] = ld_b128(kl0); bm.q[1] = ld_b128(kl0 + 8);
      v8f s = {};
      s = wmma_bf16(a0.v, bm.v, s);
      const bf16* kl1 = &ktile[keyl * KT_STRIDE + 32 + (lo ? 0 : 16)];
      bm.q[0] = ld_b128(kl1); bm.q[1] = ld_b128(kl1 + 8);
      s = wmma_bf16(a1.v, bm.v, s);

      const float m = mask[(size_t)bidx * S_ + kb + keyl];
#pragma unroll
      for (int r = 0; r < 8; ++r) {
        float p = __expf(s[r] * 0.125f) * m;      // 1/sqrt(DK)=0.125; mask over keys
        sums[r] += p;
        plds[wave][(r + (lo ? 0 : 8)) * 40 + j2 * 16 + nlane] = (bf16)p;
      }
    }

    // ---- P @ V (probs relayout through per-wave LDS; DS ops in-order per wave)
    BF16x16 pa;
    pa.q[0] = ld_b128(&plds[wave][row_a * 40 + klo]);
    pa.q[1] = ld_b128(&plds[wave][row_a * 40 + 8 + klo]);
#pragma unroll
    for (int nt = 0; nt < 4; ++nt) {
      const bf16* vl = &vtile[(nt * 16 + nlane) * VT_STRIDE + (lo ? 0 : 16)];
      BF16x16 bv_;
      bv_.q[0] = ld_b128(vl); bv_.q[1] = ld_b128(vl + 8);
      cacc[nt] = wmma_bf16(pa.v, bv_.v, cacc[nt]);
    }
    __syncthreads();   // protect ktile/vtile before next stage
  }

#pragma unroll
  for (int r = 0; r < 8; ++r) {
    float srow = sums[r];
#pragma unroll
    for (int off = 8; off >= 1; off >>= 1)
      srow += __shfl_xor(srow, off, 16);          // reduce within 16-lane half
    const float inv = 1.0f / (srow + 1e-8f);
    const int g = qbase + r + (lo ? 0 : 8);
    bf16* crow = ctx + ((size_t)bidx * S_ + g) * HD_ + h * DK_;
#pragma unroll
    for (int nt = 0; nt < 4; ++nt)
      crow[nt * 16 + nlane] = (bf16)(cacc[nt][r] * inv);
  }
}

// ---------------- output projection ----------------
// grid: B*S/64 blocks, 128 threads
__global__ void __launch_bounds__(128) out_proj(
    const bf16* __restrict__ ctx,   // [B*S, HD]
    const bf16* __restrict__ Wot,   // [D, HD] (transposed)
    const float* __restrict__ bo,
    float* __restrict__ out)        // [B*S, D]
{
  const int lane   = threadIdx.x & 31;
  const int wave   = threadIdx.x >> 5;
  const int mbase  = (blockIdx.x * 4 + wave) * 16;
  const bool lo    = lane < 16;
  const int row_a  = lane & 15;
  const int klo    = lo ? 0 : 16;
  const int kbhalf = lo ? 0 : 16;
  const int nlane  = lane & 15;

  for (int nt = 0; nt < D_ / 16; ++nt) {
    const int col = nt * 16 + nlane;
    v8f acc = {};
    for (int kc = 0; kc < HD_ / 32; ++kc) {
      const int kb = kc * 32;
      BF16x16 a;
      const bf16* arow = ctx + (size_t)(mbase + row_a) * HD_ + kb;
      a.q[0] = ld_b128(arow + klo);
      a.q[1] = ld_b128(arow + 8 + klo);
      BF16x16 b;
      const bf16* bp = Wot + (size_t)col * HD_ + kb + kbhalf;
      b.q[0] = ld_b128(bp); b.q[1] = ld_b128(bp + 8);
      acc = wmma_bf16(a.v, b.v, acc);
    }
    const float bc = bo[col];
#pragma unroll
    for (int r = 0; r < 8; ++r) {
      const int g = mbase + r + (lo ? 0 : 8);
      out[(size_t)g * D_ + col] = acc[r] + bc;
    }
  }
}

// ---------------- launch ----------------

extern "C" void kernel_launch(void* const* d_in, const int* in_sizes, int n_in,
                              void* d_out, int out_size, void* d_ws, size_t ws_size,
                              hipStream_t stream) {
  const float* Q    = (const float*)d_in[0];
  const float* mask = (const float*)d_in[1];
  const float* Wq   = (const float*)d_in[2];
  const float* bq   = (const float*)d_in[3];
  const float* Wk   = (const float*)d_in[4];
  const float* bk   = (const float*)d_in[5];
  const float* Wv   = (const float*)d_in[6];
  const float* bv   = (const float*)d_in[7];
  const float* Wo   = (const float*)d_in[8];
  const float* bo   = (const float*)d_in[9];
  float* out = (float*)d_out;

  char* ws = (char*)d_ws;
  bf16* Qb   = (bf16*)ws; ws += (size_t)B_ * S_ * D_ * sizeof(bf16);    // 8 MB
  bf16* Wqt  = (bf16*)ws; ws += (size_t)D_ * HD_ * sizeof(bf16);        // 512 KB
  bf16* Wkt  = (bf16*)ws; ws += (size_t)D_ * HD_ * sizeof(bf16);
  bf16* Wvt  = (bf16*)ws; ws += (size_t)D_ * HD_ * sizeof(bf16);
  bf16* Wot  = (bf16*)ws; ws += (size_t)HD_ * D_ * sizeof(bf16);
  bf16* qbuf = (bf16*)ws; ws += (size_t)B_ * H_ * S_ * DK_ * sizeof(bf16); // 8 MB
  bf16* kbuf = (bf16*)ws; ws += (size_t)B_ * H_ * S_ * DK_ * sizeof(bf16);
  bf16* vtw  = (bf16*)ws; ws += (size_t)B_ * H_ * DK_ * S_ * sizeof(bf16);
  bf16* ctx  = (bf16*)ws; ws += (size_t)B_ * S_ * HD_ * sizeof(bf16);

  const int nQ = B_ * S_ * D_;
  cvt_f32_bf16<<<(nQ + 255) / 256, 256, 0, stream>>>(Q, Qb, nQ);
  const int nW = D_ * HD_;
  transpose_cvt<<<(nW + 255) / 256, 256, 0, stream>>>(Wq, Wqt, D_, HD_);
  transpose_cvt<<<(nW + 255) / 256, 256, 0, stream>>>(Wk, Wkt, D_, HD_);
  transpose_cvt<<<(nW + 255) / 256, 256, 0, stream>>>(Wv, Wvt, D_, HD_);
  transpose_cvt<<<(nW + 255) / 256, 256, 0, stream>>>(Wo, Wot, HD_, D_);

  qkv_proj<<<B_ * S_ / 64, 128, 0, stream>>>(Qb, Wqt, Wkt, Wvt, bq, bk, bv,
                                             qbuf, kbuf, vtw);
  attn<<<B_ * H_ * (S_ / 64), 128, 0, stream>>>(qbuf, kbuf, vtw, mask, ctx);
  out_proj<<<B_ * S_ / 64, 128, 0, stream>>>(ctx, Wot, bo, out);
}
